// HMM_Inference_54004918780449
// MI455X (gfx1250) — compile-verified, matrix operand order checked by reading, not run on hardware
//
#include <hip/hip_runtime.h>
#include <hip/hip_bf16.h>

typedef __attribute__((ext_vector_type(16))) __bf16 v16bf;
typedef __attribute__((ext_vector_type(8)))  __bf16 v8bf;
typedef __attribute__((ext_vector_type(4)))  __bf16 v4bf;
typedef __attribute__((ext_vector_type(8)))  float  v8f;
typedef __attribute__((ext_vector_type(4)))  float  v4f;

#define S_DIM 4096
#define GWG   128     // persistent workgroups (grid barrier size)
#define RPW   32      // rows per workgroup
#define BLOCK 256

// ---------------------------------------------------------------------------
// Zero the per-step barrier counters (must be re-zeroed every launch since the
// harness does not re-poison between replays).
// ---------------------------------------------------------------------------
__global__ void hmm_zero_bar(unsigned int* __restrict__ bar, int n) {
    int i = blockIdx.x * blockDim.x + threadIdx.x;
    if (i < n) bar[i] = 0u;
}

// ---------------------------------------------------------------------------
// One-time (per launch): T_bf16[i] = exp(log_trans[i]).  64MB f32 -> 32MB bf16
// so the whole transition matrix lives in the 192MB L2 for the scan.
// ---------------------------------------------------------------------------
__global__ void hmm_exp_trans(const float* __restrict__ lt, __bf16* __restrict__ Tb,
                              long long n4) {
    long long i = (long long)blockIdx.x * blockDim.x + threadIdx.x;
    const long long stride = (long long)gridDim.x * blockDim.x;
    const v4f* src = (const v4f*)lt;
    v4bf* dst = (v4bf*)Tb;
    for (; i < n4; i += stride) {
        v4f x = src[i];
        v4bf o;
        o[0] = (__bf16)__expf(x[0]);
        o[1] = (__bf16)__expf(x[1]);
        o[2] = (__bf16)__expf(x[2]);
        o[3] = (__bf16)__expf(x[3]);
        dst[i] = o;
    }
}

// ---------------------------------------------------------------------------
// t = 0: w0 = exp(log_M0 + log_emit[0]) (unnormalized), per-WG partial sums.
// Grid = GWG blocks of 32 threads (one wave32 each), block b owns 32 states.
// ---------------------------------------------------------------------------
__global__ void hmm_init(const float* __restrict__ log_M0,
                         const float* __restrict__ emit,
                         __bf16* __restrict__ wbuf,
                         float* __restrict__ Zpart) {
    const int tid = threadIdx.x;   // 0..31 (one wave)
    const int i = blockIdx.x * 32 + tid;
    float w = __expf(log_M0[i] + emit[i]);   // emit row 0
    wbuf[i] = (__bf16)w;
    float z = w;
    #pragma unroll
    for (int s = 16; s > 0; s >>= 1) z += __shfl_down(z, s);
    if (tid == 0) Zpart[blockIdx.x] = z;
}

// ---------------------------------------------------------------------------
// Persistent scan kernel: T sequential steps, one grid barrier per step.
// Per step: v = T_bf16 @ w_{t-1} (WMMA GEMV, matrix streamed from L2),
//           w_t = v * exp(emit_t) / Z_{t-1},  Zpart[t][wg] = local sum.
// ---------------------------------------------------------------------------
__global__ __launch_bounds__(BLOCK) void hmm_scan(
    const __bf16* __restrict__ Tb,
    const float*  __restrict__ emit,
    __bf16*       __restrict__ wbuf,    // ping-pong 2 x S
    float*        __restrict__ Zpart,   // (T+1) x GWG
    unsigned int* __restrict__ bar,     // (T+1)
    int T) {
    __shared__ float part[2][4][16];    // [tile][kseg][row-in-tile]
    __shared__ float zbcast;

    const int tid  = threadIdx.x;
    const int wave = tid >> 5;
    const int lane = tid & 31;
    const int wg   = blockIdx.x;
    const int kseg = wave & 3;          // 4-way K split (1024 each)
    const int tile = wave >> 2;         // 2 row tiles of 16
    const int rbase = wg * RPW + tile * 16;
    const int row   = rbase + (lane & 15);
    // 16-bit A-fragment K mapping: lanes 0-15 -> K{0..7,16..23}, 16-31 -> +8
    const int khalf = (lane & 16) ? 8 : 0;
    const __bf16* arow = Tb + (size_t)row * S_DIM + kseg * 1024 + khalf;

    for (int t = 1; t <= T; ++t) {
        // Z_{t-1} = deterministic fixed-order sum of 128 WG partials (wave 0)
        if (wave == 0) {
            const float* zp = Zpart + (size_t)(t - 1) * GWG + lane * 4;
            float z = zp[0] + zp[1] + zp[2] + zp[3];
            #pragma unroll
            for (int s = 16; s > 0; s >>= 1) z += __shfl_down(z, s);
            if (lane == 0) zbcast = z;
        }
        // prefetch next step's emission row (HBM) while we crunch this step
        if (t < T && tid < RPW) {
            __builtin_prefetch(emit + (size_t)(t + 1) * S_DIM + wg * RPW + tid, 0, 1);
        }
        __syncthreads();

        // GEMV slice: 16 rows x 1024 K per wave, 32 chunks of 16x16x32 WMMA
        const __bf16* ap = arow;
        const __bf16* bp = wbuf + ((t - 1) & 1) * S_DIM + kseg * 1024 + khalf;
        v8f c = {0.f, 0.f, 0.f, 0.f, 0.f, 0.f, 0.f, 0.f};
        #pragma unroll 8
        for (int kc = 0; kc < 32; ++kc) {
            v8bf alo = *(const v8bf*)(ap);
            v8bf ahi = *(const v8bf*)(ap + 16);
            v8bf blo = *(const v8bf*)(bp);   // w chunk broadcast into all N cols
            v8bf bhi = *(const v8bf*)(bp + 16);
            v16bf a, b;
            #pragma unroll
            for (int i = 0; i < 8; ++i) {
                a[i] = alo[i]; a[i + 8] = ahi[i];
                b[i] = blo[i]; b[i + 8] = bhi[i];
            }
            c = __builtin_amdgcn_wmma_f32_16x16x32_bf16(
                    false, a, false, b, (short)0, c, false, false);
            ap += 32; bp += 32;
        }
        // C layout: VGPR m holds row rbase+m (lanes 0-15) / rbase+8+m (16-31);
        // all N columns equal (B broadcast) so lane 0 / lane 16 suffice.
        if ((lane & 15) == 0) {
            const int half = lane >> 4;
            #pragma unroll
            for (int m = 0; m < 8; ++m) part[tile][kseg][half * 8 + m] = c[m];
        }
        __syncthreads();

        // combine K segments, apply emission + 1/Z_{t-1}, emit w_t (bf16)
        if (tid < RPW) {
            const float invZ = __builtin_amdgcn_rcpf(zbcast);
            const int til = tid >> 4, ri = tid & 15;
            float v = part[til][0][ri] + part[til][1][ri] +
                      part[til][2][ri] + part[til][3][ri];
            const int grow = wg * RPW + tid;
            const float e = __expf(emit[(size_t)t * S_DIM + grow]);
            float w = v * e * invZ;
            wbuf[(t & 1) * S_DIM + grow] = (__bf16)w;
            float zl = w;
            #pragma unroll
            for (int s = 16; s > 0; s >>= 1) zl += __shfl_down(zl, s);
            if (tid == 0) Zpart[(size_t)t * GWG + wg] = zl;
        }
        __syncthreads();

        // device-wide barrier: release-arrive, acquire-spin, then L0 refresh
        if (tid == 0) {
            __threadfence();
            __hip_atomic_fetch_add(bar + t, 1u, __ATOMIC_RELEASE,
                                   __HIP_MEMORY_SCOPE_AGENT);
            unsigned v;
            do {
                v = __hip_atomic_load(bar + t, __ATOMIC_ACQUIRE,
                                      __HIP_MEMORY_SCOPE_AGENT);
                if (v < (unsigned)GWG) __builtin_amdgcn_s_sleep(2);
            } while (v < (unsigned)GWG);
        }
        __syncthreads();
        __threadfence();   // acquire: make remote wbuf/Zpart visible to plain loads
    }
}

// ---------------------------------------------------------------------------
// likelihood = sum_{t=0..T} log( sum_g Zpart[t][g] )  (fixed order -> deterministic)
// ---------------------------------------------------------------------------
__global__ void hmm_finish(const float* __restrict__ Zpart, float* __restrict__ out,
                           int T) {
    __shared__ double red[BLOCK];
    const int tid = threadIdx.x;
    double acc = 0.0;
    for (int t = tid; t <= T; t += BLOCK) {
        const float* zp = Zpart + (size_t)t * GWG;
        float z = 0.f;
        #pragma unroll 8
        for (int g = 0; g < GWG; ++g) z += zp[g];
        acc += (double)__logf(z);
    }
    red[tid] = acc;
    __syncthreads();
    for (int s = BLOCK / 2; s > 0; s >>= 1) {
        if (tid < s) red[tid] += red[tid + s];
        __syncthreads();
    }
    if (tid == 0) out[0] = (float)red[0];
}

// ---------------------------------------------------------------------------
extern "C" void kernel_launch(void* const* d_in, const int* in_sizes, int n_in,
                              void* d_out, int out_size, void* d_ws, size_t ws_size,
                              hipStream_t stream) {
    const float* log_M0    = (const float*)d_in[0];
    const float* log_trans = (const float*)d_in[1];
    const float* log_emit  = (const float*)d_in[2];
    const int S = S_DIM;
    const int T = in_sizes[2] / S - 1;     // log_emit is (T+1) x S

    // workspace layout
    char* ws = (char*)d_ws;
    __bf16* Tb = (__bf16*)ws;                               // S*S bf16 = 32 MB
    size_t off = (size_t)S * S * sizeof(__bf16);
    __bf16* wbuf = (__bf16*)(ws + off);                     // 2 x S bf16
    off += (size_t)2 * S * sizeof(__bf16);
    off = (off + 255) & ~(size_t)255;
    float* Zpart = (float*)(ws + off);                      // (T+1) x GWG f32
    off += (size_t)(T + 1) * GWG * sizeof(float);
    off = (off + 255) & ~(size_t)255;
    unsigned int* bar = (unsigned int*)(ws + off);          // (T+1) u32

    hmm_zero_bar<<<(T + 1 + 255) / 256, 256, 0, stream>>>(bar, T + 1);
    hmm_exp_trans<<<2048, BLOCK, 0, stream>>>(log_trans, Tb,
                                              (long long)S * S / 4);
    hmm_init<<<GWG, 32, 0, stream>>>(log_M0, log_emit, wbuf, Zpart);
    hmm_scan<<<GWG, BLOCK, 0, stream>>>(Tb, log_emit, wbuf, Zpart, bar, T);
    hmm_finish<<<1, BLOCK, 0, stream>>>(Zpart, (float*)d_out, T);
}
